// Lstm_64390149701828
// MI455X (gfx1250) — matrix operand order repaired and checked by reading, not compiled, not run on hardware
//
#include <hip/hip_runtime.h>
#include <hip/hip_bf16.h>

typedef __attribute__((ext_vector_type(16))) _Float16 v16h;
typedef __attribute__((ext_vector_type(8)))  _Float16 v8h;
typedef __attribute__((ext_vector_type(8)))  float    v8f;

#define H 128
#define T 2048
#define B 128
#define G4 512                 // 4*H gates
#define HPAD 136               // padded f16 row (16B pad -> no LDS bank clash)

// ---- fast transcendentals: gfx1250 has V_TANH_F32 (TRANS op) --------------
#if defined(__has_builtin)
#  if __has_builtin(__builtin_amdgcn_tanhf)
#    define FAST_TANH(x) __builtin_amdgcn_tanhf(x)
#  endif
#endif
#ifndef FAST_TANH
static __device__ __forceinline__ float tanh_sw(float x) {
    float e = __expf(2.0f * x);
    return (e - 1.0f) / (e + 1.0f);
}
#  define FAST_TANH(x) tanh_sw(x)
#endif
static __device__ __forceinline__ float sig_f(float x) {
    return fmaf(FAST_TANH(0.5f * x), 0.5f, 0.5f);   // sigmoid via tanh
}

static __device__ __forceinline__ v16h ld_frag(const _Float16* p0, const _Float16* p1) {
    v8h lo = *(const v8h*)p0;
    v8h hi = *(const v8h*)p1;
    return __builtin_shufflevector(lo, hi, 0,1,2,3,4,5,6,7,8,9,10,11,12,13,14,15);
}
// packed B-fragment for tile (nt,kt): lane holds 16 contiguous halves
static __device__ __forceinline__ v16h bfrag(const _Float16* Wm, int nt, int kt, int lane) {
    const _Float16* p = Wm + (((nt << 2) + kt) << 9) + lane * 16;
    return ld_frag(p, p + 8);
}

// ---------------------------------------------------------------------------
// Prep: pack Whh0 / Wih1 / Whh1 (each 512x128 f32) into f16 WMMA B-fragment
// tiles: tile (nt,kt) is 32x16 (KxN); per lane L: value
// W[nt*16 + (L%16)][kt*32 + (L/16)*8 + (e%8) + (e/8)*16].   Also b = bih+bhh, Wih0 column.
// ---------------------------------------------------------------------------
__global__ void lstm_pack_kernel(const float* __restrict__ Whh0,
                                 const float* __restrict__ Wih1,
                                 const float* __restrict__ Whh1,
                                 const float* __restrict__ bih0, const float* __restrict__ bhh0,
                                 const float* __restrict__ bih1, const float* __restrict__ bhh1,
                                 const float* __restrict__ Wih0,
                                 _Float16* __restrict__ Wp,
                                 float* __restrict__ b0, float* __restrict__ b1,
                                 float* __restrict__ w0col) {
    int idx = blockIdx.x * blockDim.x + threadIdx.x;
    const int NPACK = 3 * G4 * H;          // 196608 packed halves
    if (idx < NPACK) {
        int mat    = idx >> 16;            // 0..2
        int r      = idx & 65535;
        int tile   = r >> 9;               // nt*4 + kt
        int within = r & 511;
        int nt = tile >> 2, kt = tile & 3;
        int lane = within >> 4, e = within & 15;
        int j = nt * 16 + (lane & 15);
        int k = kt * 32 + (lane >> 4) * 8 + (e & 7) + (e >> 3) * 16;
        const float* W = (mat == 0) ? Whh0 : ((mat == 1) ? Wih1 : Whh1);
        Wp[idx] = (_Float16)W[j * H + k];
    } else if (idx < NPACK + G4) {
        int j = idx - NPACK;
        b0[j] = bih0[j] + bhh0[j];
    } else if (idx < NPACK + 2 * G4) {
        int j = idx - NPACK - G4;
        b1[j] = bih1[j] + bhh1[j];
    } else if (idx < NPACK + 3 * G4) {
        int j = idx - NPACK - 2 * G4;
        w0col[j] = Wih0[j];                // Wih0 is (512,1)
    }
}

// ---------------------------------------------------------------------------
// Fused 2-layer LSTM recurrence. 8 blocks x 256 threads (8 wave32).
// Block wg owns batch rows [wg*16, wg*16+16).
// Wave w computes gate tiles {w, w+8, w+16, w+24} == the i/f/g/o blocks for
// gate columns jj = w*16 + lane%16  ->  gates applied directly on WMMA
// accumulators; c1/c2 live in VGPRs; only h1/h2 (f16) round-trip via LDS.
// Block 7 writes y[t][:] = h2[t][row 15] (global batch index 127).
// ---------------------------------------------------------------------------
__global__ void __launch_bounds__(256)
lstm_recurrence_kernel(const float* __restrict__ x,        // (T,B)
                       const _Float16* __restrict__ Wp,    // packed 3*512*128
                       const float* __restrict__ b0g,
                       const float* __restrict__ b1g,
                       const float* __restrict__ w0g,
                       float* __restrict__ yOut) {         // (T,H)
    __shared__ __align__(16) _Float16 h1s[16][HPAD];
    __shared__ __align__(16) _Float16 h2s[16][HPAD];

    const int tid   = threadIdx.x;
    const int wg    = blockIdx.x;          // 0..7
    const int lane  = tid & 31;
    const int wave  = tid >> 5;            // 0..7
    const int mlane = lane & 15;
    const int half  = lane >> 4;
    const int jj    = wave * 16 + mlane;   // gate column owned by this lane

    // zero h state
    for (int i = tid; i < 16 * HPAD; i += 256) {
        (&h1s[0][0])[i] = (_Float16)0.0f;
        (&h2s[0][0])[i] = (_Float16)0.0f;
    }

    // loop-invariant per-lane scalars (biases + Wih0 column for column jj)
    const float bi0 = b0g[jj], bf0 = b0g[jj + 128], bg0 = b0g[jj + 256], bo0 = b0g[jj + 384];
    const float wxi = w0g[jj], wxf = w0g[jj + 128], wxg = w0g[jj + 256], wxo = w0g[jj + 384];
    const float bi1 = b1g[jj], bf1 = b1g[jj + 128], bg1 = b1g[jj + 256], bo1 = b1g[jj + 384];

    const _Float16* W0 = Wp;               // Whh0 packed
    const _Float16* W1 = Wp + G4 * H;      // Wih1 packed
    const _Float16* W2 = Wp + 2 * G4 * H;  // Whh1 packed

    float c1r[8], c2r[8];
    #pragma unroll
    for (int r = 0; r < 8; ++r) { c1r[r] = 0.0f; c2r[r] = 0.0f; }

    __syncthreads();

    for (int t = 0; t < T; ++t) {
        if (t + 1 < T) __builtin_prefetch(&x[(t + 1) * B + wg * 16], 0, 1);

        // x values for this lane's 8 rows (rows half*8 .. half*8+7)
        const float4 xa = *(const float4*)&x[t * B + wg * 16 + half * 8];
        const float4 xb = *(const float4*)&x[t * B + wg * 16 + half * 8 + 4];
        const float xq[8] = { xa.x, xa.y, xa.z, xa.w, xb.x, xb.y, xb.z, xb.w };

        // ---- read old h1 (A fragments) -----------------------------------
        v16h a1[4];
        #pragma unroll
        for (int kt = 0; kt < 4; ++kt) {
            const int c0 = kt * 32 + half * 8;
            a1[kt] = ld_frag(&h1s[mlane][c0], &h1s[mlane][c0 + 16]);
        }
        __syncthreads();   // B1: old-h1 reads + prev-step h2 stores complete

        // old h2 fragments (h2s is stable until after B2)
        v16h a3[4];
        #pragma unroll
        for (int kt = 0; kt < 4; ++kt) {
            const int c0 = kt * 32 + half * 8;
            a3[kt] = ld_frag(&h2s[mlane][c0], &h2s[mlane][c0 + 16]);
        }

        // ---- layer 0 GEMM: 4 gate-block accumulators ---------------------
        v8f ai = {}, af = {}, ag = {}, ao = {};
        #pragma unroll
        for (int kt = 0; kt < 4; ++kt) {
            ai = __builtin_amdgcn_wmma_f32_16x16x32_f16(false, a1[kt], false, bfrag(W0, wave,      kt, lane), (short)0, ai, false, false);
            af = __builtin_amdgcn_wmma_f32_16x16x32_f16(false, a1[kt], false, bfrag(W0, wave + 8,  kt, lane), (short)0, af, false, false);
            ag = __builtin_amdgcn_wmma_f32_16x16x32_f16(false, a1[kt], false, bfrag(W0, wave + 16, kt, lane), (short)0, ag, false, false);
            ao = __builtin_amdgcn_wmma_f32_16x16x32_f16(false, a1[kt], false, bfrag(W0, wave + 24, kt, lane), (short)0, ao, false, false);
        }

        // ---- layer 0 gates directly on accumulators ----------------------
        #pragma unroll
        for (int r = 0; r < 8; ++r) {
            float i_ = sig_f(ai[r] + bi0 + xq[r] * wxi);
            float f_ = sig_f(af[r] + bf0 + xq[r] * wxf);
            float g_ = FAST_TANH(ag[r] + bg0 + xq[r] * wxg);
            float o_ = sig_f(ao[r] + bo0 + xq[r] * wxo);
            float c  = fmaf(f_, c1r[r], i_ * g_);
            c1r[r]   = c;
            h1s[half * 8 + r][jj] = (_Float16)(o_ * FAST_TANH(c));
        }
        __syncthreads();   // B2: new h1 visible; old-h2 reads complete

        // new h1 fragments
        v16h a2[4];
        #pragma unroll
        for (int kt = 0; kt < 4; ++kt) {
            const int c0 = kt * 32 + half * 8;
            a2[kt] = ld_frag(&h1s[mlane][c0], &h1s[mlane][c0 + 16]);
        }

        // ---- layer 1 GEMM: h1@Wih1^T + h2@Whh1^T -------------------------
        ai = (v8f){}; af = (v8f){}; ag = (v8f){}; ao = (v8f){};
        #pragma unroll
        for (int kt = 0; kt < 4; ++kt) {
            ai = __builtin_amdgcn_wmma_f32_16x16x32_f16(false, a2[kt], false, bfrag(W1, wave,      kt, lane), (short)0, ai, false, false);
            af = __builtin_amdgcn_wmma_f32_16x16x32_f16(false, a2[kt], false, bfrag(W1, wave + 8,  kt, lane), (short)0, af, false, false);
            ag = __builtin_amdgcn_wmma_f32_16x16x32_f16(false, a2[kt], false, bfrag(W1, wave + 16, kt, lane), (short)0, ag, false, false);
            ao = __builtin_amdgcn_wmma_f32_16x16x32_f16(false, a2[kt], false, bfrag(W1, wave + 24, kt, lane), (short)0, ao, false, false);
        }
        #pragma unroll
        for (int kt = 0; kt < 4; ++kt) {
            ai = __builtin_amdgcn_wmma_f32_16x16x32_f16(false, a3[kt], false, bfrag(W2, wave,      kt, lane), (short)0, ai, false, false);
            af = __builtin_amdgcn_wmma_f32_16x16x32_f16(false, a3[kt], false, bfrag(W2, wave + 8,  kt, lane), (short)0, af, false, false);
            ag = __builtin_amdgcn_wmma_f32_16x16x32_f16(false, a3[kt], false, bfrag(W2, wave + 16, kt, lane), (short)0, ag, false, false);
            ao = __builtin_amdgcn_wmma_f32_16x16x32_f16(false, a3[kt], false, bfrag(W2, wave + 24, kt, lane), (short)0, ao, false, false);
        }

        // ---- layer 1 gates ----------------------------------------------
        #pragma unroll
        for (int r = 0; r < 8; ++r) {
            float i_ = sig_f(ai[r] + bi1);
            float f_ = sig_f(af[r] + bf1);
            float g_ = FAST_TANH(ag[r] + bg1);
            float o_ = sig_f(ao[r] + bo1);
            float c  = fmaf(f_, c2r[r], i_ * g_);
            c2r[r]   = c;
            float hv = o_ * FAST_TANH(c);
            h2s[half * 8 + r][jj] = (_Float16)hv;
            if (wg == 7 && half == 1 && r == 7) yOut[t * H + jj] = hv;  // batch 127
        }
        // no barrier here: next-iter B1 (with its dscnt-wait) closes hazards
    }
}

// ---------------------------------------------------------------------------
// Head: z = y @ fc1^T + b ; BN stats over T ; normalize+relu -> @ fc2^T + b
// ---------------------------------------------------------------------------
__global__ void fc1_kernel(const float* __restrict__ y, const float* __restrict__ w,
                           const float* __restrict__ b, float* __restrict__ z) {
    int idx = blockIdx.x * blockDim.x + threadIdx.x;   // t*128 + j
    int t = idx >> 7, j = idx & 127;
    const float* yr = y + t * H;
    const float* wr = w + j * H;
    float s = b[j];
    #pragma unroll 4
    for (int k = 0; k < H; ++k) s = fmaf(yr[k], wr[k], s);
    z[idx] = s;
}

__global__ void bnstats_kernel(const float* __restrict__ z,
                               float* __restrict__ mu, float* __restrict__ rs) {
    const int j = blockIdx.x, tid = threadIdx.x;       // 128 blocks x 256 thr
    __shared__ float sh[256], sh2[256];
    float s = 0.0f, s2 = 0.0f;
    for (int t = tid; t < T; t += 256) {
        float v = z[t * H + j];
        s += v; s2 += v * v;
    }
    sh[tid] = s; sh2[tid] = s2;
    __syncthreads();
    for (int o = 128; o > 0; o >>= 1) {
        if (tid < o) { sh[tid] += sh[tid + o]; sh2[tid] += sh2[tid + o]; }
        __syncthreads();
    }
    if (tid == 0) {
        float m   = sh[0] * (1.0f / T);
        float var = sh2[0] * (1.0f / T) - m * m;
        mu[j] = m;
        rs[j] = rsqrtf(var + 1e-5f);
    }
}

__global__ void head_kernel(const float* __restrict__ z,
                            const float* __restrict__ mu, const float* __restrict__ rs,
                            const float* __restrict__ gamma, const float* __restrict__ beta,
                            const float* __restrict__ w2, const float* __restrict__ b2,
                            float* __restrict__ out) {
    const int t = blockIdx.x, tid = threadIdx.x;       // 2048 blocks x 128 thr
    __shared__ float a[H];
    float v = (z[t * H + tid] - mu[tid]) * rs[tid] * gamma[tid] + beta[tid];
    a[tid] = v > 0.0f ? v : 0.0f;
    __syncthreads();
    if (tid < 8) {
        const float* wr = w2 + tid * H;
        float s = b2[tid];
        #pragma unroll 4
        for (int j = 0; j < H; ++j) s = fmaf(a[j], wr[j], s);
        out[t * 8 + tid] = s;
    }
}

// ---------------------------------------------------------------------------
extern "C" void kernel_launch(void* const* d_in, const int* in_sizes, int n_in,
                              void* d_out, int out_size, void* d_ws, size_t ws_size,
                              hipStream_t stream) {
    const float* x     = (const float*)d_in[0];
    const float* Wih0  = (const float*)d_in[1];
    const float* Whh0  = (const float*)d_in[2];
    const float* bih0  = (const float*)d_in[3];
    const float* bhh0  = (const float*)d_in[4];
    const float* Wih1  = (const float*)d_in[5];
    const float* Whh1  = (const float*)d_in[6];
    const float* bih1  = (const float*)d_in[7];
    const float* bhh1  = (const float*)d_in[8];
    const float* fc1w  = (const float*)d_in[9];
    const float* fc1b  = (const float*)d_in[10];
    const float* gamma = (const float*)d_in[11];
    const float* beta  = (const float*)d_in[12];
    const float* fc2w  = (const float*)d_in[13];
    const float* fc2b  = (const float*)d_in[14];
    float* out = (float*)d_out;

    // workspace carve-up
    char* ws = (char*)d_ws;
    _Float16* Wp = (_Float16*)ws;                       // 3*512*128 halves
    ws += (size_t)3 * G4 * H * sizeof(_Float16);        // 393216 B
    float* b0 = (float*)ws;  ws += G4 * sizeof(float);
    float* b1 = (float*)ws;  ws += G4 * sizeof(float);
    float* w0 = (float*)ws;  ws += G4 * sizeof(float);
    float* y  = (float*)ws;  ws += (size_t)T * H * sizeof(float);
    float* z  = (float*)ws;  ws += (size_t)T * H * sizeof(float);
    float* mu = (float*)ws;  ws += H * sizeof(float);
    float* rs = (float*)ws;

    const int packN = 3 * G4 * H + 3 * G4;              // 198144 = 774*256
    lstm_pack_kernel<<<(packN + 255) / 256, 256, 0, stream>>>(
        Whh0, Wih1, Whh1, bih0, bhh0, bih1, bhh1, Wih0, Wp, b0, b1, w0);

    lstm_recurrence_kernel<<<8, 256, 0, stream>>>(x, Wp, b0, b1, w0, y);

    fc1_kernel<<<(T * H) / 256, 256, 0, stream>>>(y, fc1w, fc1b, z);
    bnstats_kernel<<<H, 256, 0, stream>>>(z, mu, rs);
    head_kernel<<<T, H, 0, stream>>>(z, mu, rs, gamma, beta, fc2w, fc2b, out);
}
